// att_rnn_4071628997299
// MI455X (gfx1250) — matrix operand order repaired and checked by reading, not compiled, not run on hardware
//
#include <hip/hip_runtime.h>
#include <cstddef>

// ---------------------------------------------------------------------------
// Types for CDNA5 WMMA (wave32): v_wmma_f32_16x16x32_bf16
// ---------------------------------------------------------------------------
typedef __attribute__((ext_vector_type(16))) __bf16          v16bf;
typedef __attribute__((ext_vector_type(8)))  float           floatx8;
typedef __attribute__((ext_vector_type(16))) unsigned short  ushort16;
typedef __attribute__((ext_vector_type(8)))  unsigned short  ushort8;

// Native convert (v_cvt, RTNE) instead of 3-op integer rounding.
__device__ __forceinline__ unsigned short f2bf(float f) {
    __bf16 b = (__bf16)f;
    return __builtin_bit_cast(unsigned short, b);
}
__device__ __forceinline__ float bf2f(unsigned short h) {
    return __uint_as_float((unsigned)h << 16);
}
__device__ __forceinline__ float sigf(float x) { return 1.0f / (1.0f + expf(-x)); }

// A fragment (16x32 bf16, this lane's row): elems 0..15 = K[k0+hi*16 ..].
// One 32B load.
__device__ __forceinline__ v16bf load_a_frag(const unsigned short* __restrict__ row,
                                             int k0, int hi) {
    return __builtin_bit_cast(v16bf, *(const ushort16*)(row + k0 + (hi ? 16 : 0)));
}
// B fragment (32x16 bf16, this lane's column row in BT): one 32B load.
__device__ __forceinline__ v16bf load_b_frag(const unsigned short* __restrict__ row,
                                             int k0, int hi) {
    return __builtin_bit_cast(v16bf, *(const ushort16*)(row + k0 + (hi ? 16 : 0)));
}

// dot(256) of a bf16 row (32B-aligned) against an fp32 LDS vector
__device__ __forceinline__ float dot_bf16_sh(const unsigned short* __restrict__ p,
                                             const float* __restrict__ sh) {
    float s = 0.f;
    #pragma unroll 4
    for (int c = 0; c < 256; c += 8) {
        ushort8 v = *(const ushort8*)(p + c);
        #pragma unroll
        for (int j = 0; j < 8; ++j) s += sh[c + j] * bf2f(v[j]);
    }
    return s;
}

// ---------------------------------------------------------------------------
// Prep kernels
// ---------------------------------------------------------------------------
// dst[n*K+k] = bf16( src[k*N+n] (+ src2[k*N+n]) )  -- weight transpose to [N][K]
__global__ void transpose_bf_kernel(const float* __restrict__ src,
                                    const float* __restrict__ src2,
                                    unsigned short* __restrict__ dst,
                                    int K, int N) {
    int idx = blockIdx.x * blockDim.x + threadIdx.x;
    if (idx >= N * K) return;
    int n = idx / K, k = idx - n * K;
    float v = src[(size_t)k * N + n];
    if (src2) v += src2[(size_t)k * N + n];
    dst[idx] = f2bf(v);
}

__global__ void tobf_kernel(const float* __restrict__ src,
                            unsigned short* __restrict__ dst, size_t n) {
    size_t i = (size_t)blockIdx.x * blockDim.x + threadIdx.x;
    size_t stride = (size_t)gridDim.x * blockDim.x;
    for (; i < n; i += stride) dst[i] = f2bf(src[i]);
}

__global__ void init_state_kernel(const float* __restrict__ s0,
                                  const float* __restrict__ s1,
                                  unsigned short* __restrict__ hbf,
                                  float* __restrict__ cst) {
    int i = blockIdx.x * blockDim.x + threadIdx.x;
    if (i < 64 * 256) { hbf[i] = f2bf(s0[i]); cst[i] = s1[i]; }
}

// ---------------------------------------------------------------------------
// Generic WMMA GEMM: C[MxN] = A(bf16, MxK=256) @ BT(bf16, [N][256])^T + bias
// one 16x16 tile per wave, K=256 -> 8 x v_wmma_f32_16x16x32_bf16.
// Inner loop is pure b128 loads + WMMA (A pre-converted to bf16).
// ---------------------------------------------------------------------------
__global__ void gemm_bf16_kernel(const unsigned short* __restrict__ A,
                                 const unsigned short* __restrict__ BT,
                                 const float* __restrict__ bias,
                                 float* __restrict__ outF,
                                 unsigned short* __restrict__ outB,
                                 int M, int N) {
    const int lane = threadIdx.x & 31;
    const int wid  = threadIdx.x >> 5;
    const int ntiles = N >> 4;
    const int total  = (M >> 4) * ntiles;
    int w = blockIdx.x * (blockDim.x >> 5) + wid;
    if (w >= total) return;                       // wave-uniform: EXEC stays all-1
    const int mt = w / ntiles, nt = w - mt * ntiles;
    const int nn = lane & 15, hi = lane >> 4;

    const unsigned short* ap = A  + (size_t)(mt * 16 + nn) * 256; // A row = lane%16
    const unsigned short* bp = BT + (size_t)(nt * 16 + nn) * 256; // B col = lane%16

    floatx8 acc;
    float bv = bias ? bias[nt * 16 + nn] : 0.0f;
    #pragma unroll
    for (int r = 0; r < 8; ++r) acc[r] = bv;

    #pragma unroll
    for (int k0 = 0; k0 < 256; k0 += 32) {
        acc = __builtin_amdgcn_wmma_f32_16x16x32_bf16(
                  false, load_a_frag(ap, k0, hi),
                  false, load_b_frag(bp, k0, hi),
                  (short)0, acc, false, false);
    }
    #pragma unroll
    for (int r = 0; r < 8; ++r) {
        int m = mt * 16 + hi * 8 + r;             // D: vgpr r -> M = r (+8 for hi lanes)
        int n = nt * 16 + nn;
        size_t idx = (size_t)m * N + n;
        if (outF) outF[idx] = acc[r];
        if (outB) outB[idx] = f2bf(acc[r]);
    }
}

// ---------------------------------------------------------------------------
// LSTM cell 1: z = xw1[:,t,:] (incl. b1) + h_state @ U1 ; gates -> h1(bf16), c1
// 64 waves; each wave: one 16(b)x16(u) tile, 4 gate accumulators, A reused 4x.
// ---------------------------------------------------------------------------
__global__ void lstm1_kernel(const unsigned short* __restrict__ hbf,   // [64][256] bf16
                             const unsigned short* __restrict__ U1T,   // [1024][256] bf16
                             const float* __restrict__ xw1,            // [(b*128+t)][1024]
                             const float* __restrict__ cstate,         // [64][256]
                             float* __restrict__ c1,
                             unsigned short* __restrict__ h1bf,
                             int t) {
    const int lane = threadIdx.x & 31, wid = threadIdx.x >> 5;
    const int w = blockIdx.x * 8 + wid;           // 0..63
    const int bt = w >> 4, ut = w & 15;
    const int nn = lane & 15, hi = lane >> 4;

    floatx8 acc[4];
    #pragma unroll
    for (int g = 0; g < 4; ++g)
        #pragma unroll
        for (int r = 0; r < 8; ++r) {
            int m = bt * 16 + hi * 8 + r;
            acc[g][r] = xw1[((size_t)m * 128 + t) * 1024 + g * 256 + ut * 16 + nn];
        }

    const unsigned short* ap = hbf + (size_t)(bt * 16 + nn) * 256;
    #pragma unroll
    for (int k0 = 0; k0 < 256; k0 += 32) {
        v16bf a = load_a_frag(ap, k0, hi);
        #pragma unroll
        for (int g = 0; g < 4; ++g) {
            int n = g * 256 + ut * 16 + nn;
            acc[g] = __builtin_amdgcn_wmma_f32_16x16x32_bf16(
                         false, a,
                         false, load_b_frag(U1T + (size_t)n * 256, k0, hi),
                         (short)0, acc[g], false, false);
        }
    }
    #pragma unroll
    for (int r = 0; r < 8; ++r) {
        int m = bt * 16 + hi * 8 + r, u = ut * 16 + nn;
        float cs = cstate[m * 256 + u];
        float ig = sigf(acc[0][r]), fg = sigf(acc[1][r]);
        float gg = tanhf(acc[2][r]), og = sigf(acc[3][r]);
        float cv = fg * cs + ig * gg;
        c1[m * 256 + u] = cv;
        h1bf[m * 256 + u] = f2bf(og * tanhf(cv));
    }
}

// ---------------------------------------------------------------------------
// LSTM cell 2 (x=h=h1): z = h1 @ (W2+U2) + b2 ; gates with c1 -> h2
// writes h2 to out[:, t, 0:256] and to h_state (bf16) for next step / attention
// ---------------------------------------------------------------------------
__global__ void lstm2_kernel(const unsigned short* __restrict__ h1bf,
                             const unsigned short* __restrict__ W2pT,  // [1024][256]
                             const float* __restrict__ b2,
                             const float* __restrict__ c1,
                             float* __restrict__ out,                  // [64][128][512]
                             unsigned short* __restrict__ hstbf,
                             int t) {
    const int lane = threadIdx.x & 31, wid = threadIdx.x >> 5;
    const int w = blockIdx.x * 8 + wid;
    const int bt = w >> 4, ut = w & 15;
    const int nn = lane & 15, hi = lane >> 4;

    floatx8 acc[4];
    #pragma unroll
    for (int g = 0; g < 4; ++g) {
        float bv = b2[g * 256 + ut * 16 + nn];
        #pragma unroll
        for (int r = 0; r < 8; ++r) acc[g][r] = bv;
    }

    const unsigned short* ap = h1bf + (size_t)(bt * 16 + nn) * 256;
    #pragma unroll
    for (int k0 = 0; k0 < 256; k0 += 32) {
        v16bf a = load_a_frag(ap, k0, hi);
        #pragma unroll
        for (int g = 0; g < 4; ++g) {
            int n = g * 256 + ut * 16 + nn;
            acc[g] = __builtin_amdgcn_wmma_f32_16x16x32_bf16(
                         false, a,
                         false, load_b_frag(W2pT + (size_t)n * 256, k0, hi),
                         (short)0, acc[g], false, false);
        }
    }
    #pragma unroll
    for (int r = 0; r < 8; ++r) {
        int m = bt * 16 + hi * 8 + r, u = ut * 16 + nn;
        float cs = c1[m * 256 + u];
        float ig = sigf(acc[0][r]), fg = sigf(acc[1][r]);
        float gg = tanhf(acc[2][r]), og = sigf(acc[3][r]);
        float cv = fg * cs + ig * gg;
        float hv = og * tanhf(cv);
        out[((size_t)m * 128 + t) * 512 + u] = hv;     // out[:, t, :256] = h2
        hstbf[m * 256 + u] = f2bf(hv);                 // state for attn + next step
    }
}

// ---------------------------------------------------------------------------
// Fused attention per batch row b (64 WGs x 256 threads):
//   s_fi = h2@Ws+bs ; e = <s_fi, h_psi[b,te,:]> ; softmax ; ctx = alpha@h_enc[b]
// writes out[b,t,256:512] = ctx and cstate = ctx (state fed back is [h2, ctx])
// h_psi/h_enc are bf16 and L2-resident (64 MB << 192 MB L2).
// ---------------------------------------------------------------------------
__global__ void attn_kernel(const unsigned short* __restrict__ h2bf,
                            const unsigned short* __restrict__ WsT,     // [256][256]
                            const float* __restrict__ bs,
                            const unsigned short* __restrict__ hpsibf,  // [64][1024][256]
                            const unsigned short* __restrict__ hencbf,  // [64][1024][256]
                            float* __restrict__ out,
                            float* __restrict__ cstate,
                            int t) {
    __shared__ float sh_h2[256];
    __shared__ float sh_sfi[256];
    __shared__ float sh_e[1024];
    __shared__ float sh_red[256];
    const int b = blockIdx.x, tid = threadIdx.x;

    sh_h2[tid] = bf2f(h2bf[b * 256 + tid]);
    __syncthreads();

    sh_sfi[tid] = bs[tid] + dot_bf16_sh(WsT + (size_t)tid * 256, sh_h2);
    __syncthreads();

    float ev[4];
    float lmax = -3.4e38f;
    #pragma unroll
    for (int j = 0; j < 4; ++j) {
        int te = tid + j * 256;
        float e = dot_bf16_sh(hpsibf + ((size_t)b * 1024 + te) * 256, sh_sfi);
        ev[j] = e;
        lmax = fmaxf(lmax, e);
    }
    sh_red[tid] = lmax;
    __syncthreads();
    for (int s = 128; s > 0; s >>= 1) {
        if (tid < s) sh_red[tid] = fmaxf(sh_red[tid], sh_red[tid + s]);
        __syncthreads();
    }
    float mx = sh_red[0];
    __syncthreads();

    float lsum = 0.f;
    #pragma unroll
    for (int j = 0; j < 4; ++j) {
        float ex = expf(ev[j] - mx);
        sh_e[tid + j * 256] = ex;
        lsum += ex;
    }
    sh_red[tid] = lsum;
    __syncthreads();
    for (int s = 128; s > 0; s >>= 1) {
        if (tid < s) sh_red[tid] += sh_red[tid + s];
        __syncthreads();
    }
    float inv = 1.0f / sh_red[0];

    // ctx[u=tid]: block-coalesced sweep over h_enc rows (512 B per iteration)
    const unsigned short* he = hencbf + (size_t)b * 1024 * 256 + tid;
    float acc = 0.f;
    for (int te = 0; te < 1024; ++te) acc += sh_e[te] * bf2f(he[(size_t)te * 256]);
    acc *= inv;
    out[((size_t)b * 128 + t) * 512 + 256 + tid] = acc;
    cstate[b * 256 + tid] = acc;   // c state for next step = ctx (per reference)
}

// ---------------------------------------------------------------------------
// Host driver: all launches on `stream`, deterministic, graph-capturable.
// Workspace plan (~107 MB, each region 256B aligned, fully rewritten per call).
// ---------------------------------------------------------------------------
extern "C" void kernel_launch(void* const* d_in, const int* in_sizes, int n_in,
                              void* d_out, int out_size, void* d_ws, size_t ws_size,
                              hipStream_t stream) {
    (void)in_sizes; (void)n_in; (void)out_size; (void)ws_size;
    const float* x    = (const float*)d_in[0];   // [64][128][256]
    const float* s0   = (const float*)d_in[1];   // [64][256]
    const float* s1   = (const float*)d_in[2];
    const float* henc = (const float*)d_in[3];   // [64][1024][256]
    const float* W1   = (const float*)d_in[4];   // [256][1024]
    const float* U1   = (const float*)d_in[5];
    const float* b1   = (const float*)d_in[6];
    const float* W2   = (const float*)d_in[7];
    const float* U2   = (const float*)d_in[8];
    const float* b2   = (const float*)d_in[9];
    const float* Ws   = (const float*)d_in[10];  // [256][256]
    const float* bs   = (const float*)d_in[11];
    const float* Wh   = (const float*)d_in[12];
    const float* bh   = (const float*)d_in[13];
    float* out = (float*)d_out;                  // [64][128][512]

    char* ws = (char*)d_ws;
    size_t off = 0;
    auto alloc = [&](size_t bytes) -> char* {
        char* p = ws + off;
        off += (bytes + 255) & ~(size_t)255;
        return p;
    };
    unsigned short* hencbf = (unsigned short*)alloc(64ull * 1024 * 256 * 2); // 32 MB
    unsigned short* hpsibf = (unsigned short*)alloc(64ull * 1024 * 256 * 2); // 32 MB
    float*          xw1    = (float*)         alloc(64ull * 128 * 1024 * 4); // 32 MB
    unsigned short* xbf    = (unsigned short*)alloc(64ull * 128 * 256 * 2);  // 4 MB
    unsigned short* U1T    = (unsigned short*)alloc(1024ull * 256 * 2);
    unsigned short* W2pT   = (unsigned short*)alloc(1024ull * 256 * 2);
    unsigned short* W1T    = (unsigned short*)alloc(1024ull * 256 * 2);
    unsigned short* WsT    = (unsigned short*)alloc(256ull * 256 * 2);
    unsigned short* WhT    = (unsigned short*)alloc(256ull * 256 * 2);
    unsigned short* hstbf  = (unsigned short*)alloc(64ull * 256 * 2);
    unsigned short* h1bf   = (unsigned short*)alloc(64ull * 256 * 2);
    float*          c1     = (float*)         alloc(64ull * 256 * 4);
    float*          cstate = (float*)         alloc(64ull * 256 * 4);

    // ---- one-time prep (per call; deterministic) ----
    transpose_bf_kernel<<<1024, 256, 0, stream>>>(U1, nullptr, U1T, 256, 1024);
    transpose_bf_kernel<<<1024, 256, 0, stream>>>(W2, U2,      W2pT, 256, 1024); // W2+U2
    transpose_bf_kernel<<<1024, 256, 0, stream>>>(W1, nullptr, W1T, 256, 1024);
    transpose_bf_kernel<<< 256, 256, 0, stream>>>(Ws, nullptr, WsT, 256, 256);
    transpose_bf_kernel<<< 256, 256, 0, stream>>>(Wh, nullptr, WhT, 256, 256);
    tobf_kernel<<<2048, 256, 0, stream>>>(henc, hencbf, 64ull * 1024 * 256);
    tobf_kernel<<< 512, 256, 0, stream>>>(x, xbf, 64ull * 128 * 256);
    init_state_kernel<<<64, 256, 0, stream>>>(s0, s1, hstbf, cstate);

    // h_psi = h_enc @ Wh + bh   (65536 x 256, bf16 out); A = bf16 h_enc copy
    gemm_bf16_kernel<<<8192, 256, 0, stream>>>(hencbf, WhT, bh, nullptr, hpsibf, 65536, 256);
    // xw1 = x @ W1 + b1         (8192 x 1024, fp32 out, rows = b*128+t)
    gemm_bf16_kernel<<<4096, 256, 0, stream>>>(xbf, W1T, b1, xw1, nullptr, 8192, 1024);

    // ---- sequential decoder scan: 3 kernels per step ----
    for (int t = 0; t < 128; ++t) {
        lstm1_kernel<<<8, 256, 0, stream>>>(hstbf, U1T, xw1, cstate, c1, h1bf, t);
        lstm2_kernel<<<8, 256, 0, stream>>>(h1bf, W2pT, b2, c1, out, hstbf, t);
        attn_kernel<<<64, 256, 0, stream>>>(hstbf, WsT, bs, hpsibf, hencbf, out, cstate, t);
    }
}
